// GaussianGeoNetwork_90615220011503
// MI455X (gfx1250) — matrix operand order repaired and checked by reading, not compiled, not run on hardware
//
#include <hip/hip_runtime.h>
#include <hip/hip_bf16.h>

typedef __attribute__((ext_vector_type(16))) _Float16 v16h;
typedef __attribute__((ext_vector_type(8)))  _Float16 v8h;
typedef __attribute__((ext_vector_type(8)))  float    v8f;

#define WAVES 4
#define TPW 8                         // tiles (of 16 faces) per wave
#define FACES_PER_WG (WAVES * TPW * 16)
#define KPAD 288                      // 264 input features padded to 9 chunks of 32

__device__ __forceinline__ float sigm(float x) { return 1.0f / (1.0f + __expf(-x)); }

__device__ __forceinline__ v8f wmma16(v16h a, v16h b, v8f c) {
  // v_wmma_f32_16x16x32_f16 : D = A(16x32 f16) * B(32x16 f16) + C(16x16 f32)
  return __builtin_amdgcn_wmma_f32_16x16x32_f16(false, a, false, b, (short)0, c, false, false);
}

// A-fragment: lane holds row m = lane%16; elements 0..7 -> K = k0+8*half+e,
// elements 8..15 -> K = k0+16+8*half+(e-8). Two contiguous 16B LDS loads.
__device__ __forceinline__ v16h load_a_frag(const _Float16* base, int row, int rowStride,
                                            int k0, int half) {
  union { v16h v; struct { v8h lo; v8h hi; } s; } u;
  const _Float16* p = base + row * rowStride + k0 + 8 * half;
  u.s.lo = *(const v8h*)p;
  u.s.hi = *(const v8h*)(p + 16);
  return u.v;
}

// B-fragment from an [n][k] (pre-transposed) LDS tile: lane holds column
// n = col; elements 0..7 -> K = k0+8*half+e, elements 8..15 -> K = k0+16+8*half+(e-8).
__device__ __forceinline__ v16h load_b_frag(const _Float16* base, int col, int rowStride,
                                            int k0, int half) {
  union { v16h v; struct { v8h lo; v8h hi; } s; } u;
  const _Float16* p = base + col * rowStride + k0 + 8 * half;
  u.s.lo = *(const v8h*)p;
  u.s.hi = *(const v8h*)(p + 16);
  return u.v;
}

__global__ __launch_bounds__(WAVES * 32)
void GaussianGeoNetwork_90615220011503_kernel(
    const float* __restrict__ embeded, const float* __restrict__ vertices,
    const int*   __restrict__ faces,
    const float* __restrict__ W1, const float* __restrict__ B1,
    const float* __restrict__ W2, const float* __restrict__ B2,
    const float* __restrict__ Wb, const float* __restrict__ bb,
    const float* __restrict__ Wo, const float* __restrict__ bo,
    const float* __restrict__ Ws, const float* __restrict__ bs,
    const float* __restrict__ Wr, const float* __restrict__ br,
    const float* __restrict__ Wg, const float* __restrict__ bg,
    float* __restrict__ out, int NF) {
  // ---- LDS (~111 KB; CDNA5 allows up to 320 KB per workgroup) ----
  __shared__ __align__(32) _Float16 sW1t[64 * KPAD];   // W1^T  [n][k], k padded
  __shared__ __align__(32) _Float16 sW2t[64 * 64];     // W2^T  [n][k]
  __shared__ __align__(32) _Float16 sWht[32 * 64];     // heads^T [n][k], 24 used
  __shared__ float sB1[64], sB2[64], sBh[32];
  __shared__ __align__(32) _Float16 sA [WAVES][16 * KPAD];  // activations (16 faces x K)
  __shared__ __align__(32) _Float16 sH1[WAVES][16 * 64];
  __shared__ __align__(32) _Float16 sH2[WAVES][16 * 64];
  __shared__ float sHead[WAVES][16 * 32];
  __shared__ float sGeom[WAVES][16 * 8];               // v0.xyz, normal.xyz per face
  __shared__ int   sVid [WAVES][16 * 3];

  const int tid  = threadIdx.x;
  const int w    = tid >> 5;
  const int lane = tid & 31;
  const int m16  = lane & 15;
  const int half = lane >> 4;
  const int ntiles = (NF + 15) >> 4;

  // ---- stage weights (f32 -> f16, transposed to [n][k]) ----
  for (int i = tid; i < 64 * KPAD; i += WAVES * 32) {
    int n = i / KPAD, k = i - n * KPAD;
    sW1t[i] = (k < 264) ? (_Float16)W1[k * 64 + n] : (_Float16)0.0f;
  }
  for (int i = tid; i < 64 * 64; i += WAVES * 32) {
    int n = i >> 6, k = i & 63;
    sW2t[i] = (_Float16)W2[k * 64 + n];
  }
  for (int i = tid; i < 32 * 64; i += WAVES * 32) {
    int n = i >> 6, k = i & 63;
    float v;
    if (n < 3)       v = Wb[k * 3 + n];
    else if (n == 3) v = Wo[k];
    else if (n < 7)  v = Ws[k * 3 + (n - 4)];
    else if (n == 7) v = Wr[k];
    else if (n < 24) v = Wg[k * 16 + (n - 8)];
    else             v = 0.0f;
    sWht[i] = (_Float16)v;
  }
  if (tid < 64) { sB1[tid] = B1[tid]; sB2[tid] = B2[tid]; }
  if (tid < 32) {
    int n = tid; float v;
    if (n < 3)       v = bb[n];
    else if (n == 3) v = bo[0];
    else if (n < 7)  v = bs[n - 4];
    else if (n == 7) v = br[0];
    else if (n < 24) v = bg[n - 8];
    else             v = 0.0f;
    sBh[n] = v;
  }
  // zero the K padding of this wave's A tile once (cols 264..287 stay 0)
  for (int i = lane; i < 16 * 24; i += 32) {
    int f = i / 24, c = i - f * 24;
    sA[w][f * KPAD + 264 + c] = (_Float16)0.0f;
  }
  __syncthreads();

  const long oOp  = 3L  * NF;
  const long oSc  = 4L  * NF;
  const long oRot = 7L  * NF;
  const long oGeo = 16L * NF;

  for (int t = 0; t < TPW; ++t) {
    const int tile  = (blockIdx.x * WAVES + w) * TPW + t;
    const bool valid = tile < ntiles;           // wave-uniform
    const int fbase = tile * 16;

    // ---- gather: face ids, normals, embeddings -> A tile (f16) ----
    if (valid) {
      if (lane < 16) {
        int face = fbase + lane;
        int i0 = faces[face * 3 + 0];
        int i1 = faces[face * 3 + 1];
        int i2 = faces[face * 3 + 2];
        sVid[w][lane * 3 + 0] = i0;
        sVid[w][lane * 3 + 1] = i1;
        sVid[w][lane * 3 + 2] = i2;
        float ax = vertices[i0 * 3 + 0], ay = vertices[i0 * 3 + 1], az = vertices[i0 * 3 + 2];
        float bx = vertices[i1 * 3 + 0], by = vertices[i1 * 3 + 1], bz = vertices[i1 * 3 + 2];
        float cx = vertices[i2 * 3 + 0], cy = vertices[i2 * 3 + 1], cz = vertices[i2 * 3 + 2];
        float e1x = ax - bx, e1y = ay - by, e1z = az - bz;
        float e2x = ax - cx, e2y = ay - cy, e2z = az - cz;
        float nx = e1y * e2z - e1z * e2y;
        float ny = e1z * e2x - e1x * e2z;
        float nz = e1x * e2y - e1y * e2x;
        float nrm = sqrtf(nx * nx + ny * ny + nz * nz);
        float inv = 1.0f / fmaxf(nrm, 1e-12f);
        nx *= inv; ny *= inv; nz *= inv;
        sGeom[w][lane * 8 + 0] = ax; sGeom[w][lane * 8 + 1] = ay; sGeom[w][lane * 8 + 2] = az;
        sGeom[w][lane * 8 + 3] = nx; sGeom[w][lane * 8 + 4] = ny; sGeom[w][lane * 8 + 5] = nz;
        sA[w][lane * KPAD + 261] = (_Float16)nx;
        sA[w][lane * KPAD + 262] = (_Float16)ny;
        sA[w][lane * KPAD + 263] = (_Float16)nz;
      }
      __builtin_amdgcn_wave_barrier();
      // coalesced, division-free embedding gather: consecutive lanes read
      // consecutive floats of one embedding row (128B/wave per access)
      for (int f = 0; f < 16; ++f) {
        const int base0 = sVid[w][f * 3 + 0] * 87;
        const int base1 = sVid[w][f * 3 + 1] * 87;
        const int base2 = sVid[w][f * 3 + 2] * 87;
        _Float16* arow = &sA[w][f * KPAD];
        #pragma unroll
        for (int j0 = 0; j0 < 87; j0 += 32) {
          int j = j0 + lane;
          if (j < 87) {
            arow[j]        = (_Float16)embeded[base0 + j];
            arow[87 + j]   = (_Float16)embeded[base1 + j];
            arow[174 + j]  = (_Float16)embeded[base2 + j];
          }
        }
      }
    }
    __syncthreads();

    // ---- GEMM1: h1(16x64) = A(16x288) @ W1(288x64), +b1, relu ----
    if (valid) {
      v16h afr[9];
      #pragma unroll
      for (int kc = 0; kc < 9; ++kc)
        afr[kc] = load_a_frag(&sA[w][0], m16, KPAD, kc * 32, half);
      for (int nt = 0; nt < 4; ++nt) {
        v8f acc = {};
        #pragma unroll
        for (int kc = 0; kc < 9; ++kc) {
          v16h b = load_b_frag(sW1t, nt * 16 + m16, KPAD, kc * 32, half);
          acc = wmma16(afr[kc], b, acc);
        }
        float bias = sB1[nt * 16 + m16];
        #pragma unroll
        for (int r = 0; r < 8; ++r) {
          float hv = acc[r] + bias;
          hv = hv > 0.0f ? hv : 0.0f;
          sH1[w][(r + 8 * half) * 64 + nt * 16 + m16] = (_Float16)hv;
        }
      }
    }
    __syncthreads();

    // ---- GEMM2: h2 = relu(h1 @ W2 + b2) ----
    if (valid) {
      v16h afr[2];
      #pragma unroll
      for (int kc = 0; kc < 2; ++kc)
        afr[kc] = load_a_frag(&sH1[w][0], m16, 64, kc * 32, half);
      for (int nt = 0; nt < 4; ++nt) {
        v8f acc = {};
        #pragma unroll
        for (int kc = 0; kc < 2; ++kc) {
          v16h b = load_b_frag(sW2t, nt * 16 + m16, 64, kc * 32, half);
          acc = wmma16(afr[kc], b, acc);
        }
        float bias = sB2[nt * 16 + m16];
        #pragma unroll
        for (int r = 0; r < 8; ++r) {
          float hv = acc[r] + bias;
          hv = hv > 0.0f ? hv : 0.0f;
          sH2[w][(r + 8 * half) * 64 + nt * 16 + m16] = (_Float16)hv;
        }
      }
    }
    __syncthreads();

    // ---- GEMM3: heads(16x24, padded 32) = h2 @ Whead + bhead ----
    if (valid) {
      v16h afr[2];
      #pragma unroll
      for (int kc = 0; kc < 2; ++kc)
        afr[kc] = load_a_frag(&sH2[w][0], m16, 64, kc * 32, half);
      for (int nt = 0; nt < 2; ++nt) {
        v8f acc = {};
        #pragma unroll
        for (int kc = 0; kc < 2; ++kc) {
          v16h b = load_b_frag(sWht, nt * 16 + m16, 64, kc * 32, half);
          acc = wmma16(afr[kc], b, acc);
        }
        float bias = sBh[nt * 16 + m16];
        #pragma unroll
        for (int r = 0; r < 8; ++r)
          sHead[w][(r + 8 * half) * 32 + nt * 16 + m16] = acc[r] + bias;
      }
    }
    __syncthreads();

    // ---- post-process + outputs ----
    if (valid) {
      const float* hd = &sHead[w][m16 * 32];
      long face = fbase + m16;
      if (half == 0) {
        // bary (normalized; reference multiplies v0 by each component)
        float b0 = sigm(hd[0]), b1v = sigm(hd[1]), b2v = sigm(hd[2]);
        float s  = b0 + b1v + b2v;
        float bsum = (b0 + b1v + b2v) / s;   // == 1, computed faithfully
        float v0x = sGeom[w][m16 * 8 + 0], v0y = sGeom[w][m16 * 8 + 1], v0z = sGeom[w][m16 * 8 + 2];
        out[face * 3 + 0] = v0x * bsum;
        out[face * 3 + 1] = v0y * bsum;
        out[face * 3 + 2] = v0z * bsum;
        out[oOp + face] = sigm(hd[3]);
        out[oSc + face * 3 + 0] = sigm(hd[4]) * 0.02f;
        out[oSc + face * 3 + 1] = sigm(hd[5]) * 0.02f;
        out[oSc + face * 3 + 2] = sigm(hd[6]) * 0.02f;
        float theta = sigm(hd[7]) * 6.283185307179586f;
        float wx = sGeom[w][m16 * 8 + 3], wy = sGeom[w][m16 * 8 + 4], wz = sGeom[w][m16 * 8 + 5];
        float sn = __sinf(theta), cs = __cosf(theta), oc = 1.0f - cs;
        float* R = &out[oRot + face * 9];
        R[0] = 1.0f + oc * (-(wz * wz + wy * wy));
        R[1] = -sn * wz + oc * (wx * wy);
        R[2] =  sn * wy + oc * (wx * wz);
        R[3] =  sn * wz + oc * (wx * wy);
        R[4] = 1.0f + oc * (-(wz * wz + wx * wx));
        R[5] = -sn * wx + oc * (wy * wz);
        R[6] = -sn * wy + oc * (wx * wz);
        R[7] =  sn * wx + oc * (wy * wz);
        R[8] = 1.0f + oc * (-(wx * wx + wy * wy));
      } else {
        #pragma unroll
        for (int j = 0; j < 16; ++j) {
          float g = hd[8 + j];
          out[oGeo + face * 16 + j] = g > 0.0f ? g : 0.0f;
        }
      }
    }
    __syncthreads();
  }
}

extern "C" void kernel_launch(void* const* d_in, const int* in_sizes, int n_in,
                              void* d_out, int out_size, void* d_ws, size_t ws_size,
                              hipStream_t stream) {
  (void)n_in; (void)out_size; (void)d_ws; (void)ws_size;
  const float* embeded  = (const float*)d_in[0];
  const float* vertices = (const float*)d_in[1];
  const int*   faces    = (const int*)d_in[2];
  const float* W1 = (const float*)d_in[3];
  const float* b1 = (const float*)d_in[4];
  const float* W2 = (const float*)d_in[5];
  const float* b2 = (const float*)d_in[6];
  const float* Wb = (const float*)d_in[7];
  const float* bb = (const float*)d_in[8];
  const float* Wo = (const float*)d_in[9];
  const float* bo = (const float*)d_in[10];
  const float* Ws = (const float*)d_in[11];
  const float* bs = (const float*)d_in[12];
  const float* Wr = (const float*)d_in[13];
  const float* br = (const float*)d_in[14];
  const float* Wg = (const float*)d_in[15];
  const float* bg = (const float*)d_in[16];
  float* out = (float*)d_out;

  const int NF = in_sizes[2] / 3;
  const int grid = (NF + FACES_PER_WG - 1) / FACES_PER_WG;
  GaussianGeoNetwork_90615220011503_kernel<<<dim3(grid), dim3(WAVES * 32), 0, stream>>>(
      embeded, vertices, faces, W1, b1, W2, b2, Wb, bb, Wo, bo, Ws, bs, Wr, br, Wg, bg,
      out, NF);
}
